// PredNetNoMap_19576460935605
// MI455X (gfx1250) — matrix operand order repaired and checked by reading, not compiled, not run on hardware
//
#include <hip/hip_runtime.h>
#include <hip/hip_bf16.h>

typedef __attribute__((ext_vector_type(16))) __bf16 v16bf;
typedef __attribute__((ext_vector_type(8)))  __bf16 v8bf;
typedef __attribute__((ext_vector_type(8)))  float  v8f;

#define DM 128      // feature dim
#define KM 6        // num modes
#define PO 60       // 2 * P
#define GN_EPS 1e-5f

// ---------------------------------------------------------------------------
// WMMA helpers (CDNA5: V_WMMA_F32_16X16X32_BF16, wave32)
// ---------------------------------------------------------------------------

// A fragment (16x32 bf16) from an LDS tile stored row-major [16][ldA] bf16.
// ISA 7.12.2 16-bit A layout: lane L (L<16) row M=L holds K=0..7,16..23;
// lane L+16 holds K=8..15,24..31 (per 32-wide k-step s).
__device__ __forceinline__ v16bf load_a_frag(const __bf16* A, int ldA, int lane, int s) {
  const int row  = lane & 15;
  const int half = lane >> 4;
  const __bf16* p = A + row * ldA + s * 32 + 8 * half;
  v8bf lo = *(const v8bf*)p;          // K = s*32 + 8*half + [0..7]
  v8bf hi = *(const v8bf*)(p + 16);   // K = s*32 + 16 + 8*half + [0..7]
  v16bf r;
#pragma unroll
  for (int i = 0; i < 8; ++i) { r[i] = lo[i]; r[i + 8] = hi[i]; }
  return r;
}

__device__ __forceinline__ v8f wmma_bf16(v16bf a, v16bf b, v8f c) {
  return __builtin_amdgcn_wmma_f32_16x16x32_bf16(false, a, false, b, (short)0, c,
                                                 false, false);
}

// B fragments are pre-packed by pack_kernel so lane's 16 bf16 are contiguous:
// packed[((ct*ksteps + s)*32 + lane)*16 + e]  with k = s*32 + 8*(lane>=16) + (e&7) + 16*(e>>3),
// col = ct*16 + (lane&15).
template<int KSTEPS>
__device__ __forceinline__ v8f gemm_col(const __bf16* A, int ldA,
                                        const v16bf* __restrict__ Wp,
                                        int c, int lane, v8f acc) {
#pragma unroll
  for (int s = 0; s < KSTEPS; ++s) {
    v16bf a = load_a_frag(A, ldA, lane, s);
    v16bf b = Wp[(c * KSTEPS + s) * 32 + lane];
    acc = wmma_bf16(a, b, acc);
  }
  return acc;
}

// D tile (f32 16x16): lane -> col = lane&15, rows = 8*(lane>>4) + i.
__device__ __forceinline__ void store_tile(float* H, int ldH, int c, int lane, v8f acc) {
  const int col = c * 16 + (lane & 15);
  const int r0  = 8 * (lane >> 4);
#pragma unroll
  for (int i = 0; i < 8; ++i) H[(r0 + i) * ldH + col] = acc[i];
}

// ---------------------------------------------------------------------------
// GroupNorm(1 group over 128 ch) on a 16x128 f32 LDS tile, one wave.
// Each lane owns 4 channels; row mean/var via __shfl_xor tree.
// ---------------------------------------------------------------------------
template<bool RES, bool RELU, bool OUTB, bool OUTF>
__device__ __forceinline__ void group_norm16(const float* __restrict__ H,
                                             const float* __restrict__ g,
                                             const float* __restrict__ b,
                                             const float* __restrict__ resid,
                                             __bf16* outB, float* outF, int lane) {
  const int c0 = lane * 4;
  float gv[4], bv[4];
#pragma unroll
  for (int j = 0; j < 4; ++j) { gv[j] = g[c0 + j]; bv[j] = b[c0 + j]; }
#pragma unroll 1
  for (int r = 0; r < 16; ++r) {
    float4 xv = *(const float4*)(H + r * DM + c0);
    float x[4] = {xv.x, xv.y, xv.z, xv.w};
    float s  = x[0] + x[1] + x[2] + x[3];
    float ss = x[0]*x[0] + x[1]*x[1] + x[2]*x[2] + x[3]*x[3];
#pragma unroll
    for (int m = 16; m >= 1; m >>= 1) {
      s  += __shfl_xor(s,  m, 32);
      ss += __shfl_xor(ss, m, 32);
    }
    float mean = s * (1.f / 128.f);
    float var  = ss * (1.f / 128.f) - mean * mean;
    float inv  = rsqrtf(var + GN_EPS);
#pragma unroll
    for (int j = 0; j < 4; ++j) {
      float y = (x[j] - mean) * inv * gv[j] + bv[j];
      if (RES)  y += resid[r * DM + c0 + j];
      if (RELU) y = fmaxf(y, 0.f);
      if (OUTF) outF[r * DM + c0 + j] = y;
      if (OUTB) outB[r * DM + c0 + j] = (__bf16)y;
    }
  }
}

// Load a 16x128 f32 tile from global to LDS (f32 and/or bf16 copies).
template<bool OF, bool OB>
__device__ __forceinline__ void load_act_tile(const float* __restrict__ a, int n0,
                                              int lane, float* outF, __bf16* outB) {
#pragma unroll 1
  for (int i = 0; i < 16; ++i) {
    int t = lane + 32 * i;            // 0..511 float4s
    int r = t >> 5;
    int c = (t & 31) * 4;
    float4 v = *(const float4*)(a + (size_t)(n0 + r) * DM + c);
    if (OF) { float* o = outF + r * DM + c;
              o[0] = v.x; o[1] = v.y; o[2] = v.z; o[3] = v.w; }
    if (OB) { __bf16* o = outB + r * DM + c;
              o[0] = (__bf16)v.x; o[1] = (__bf16)v.y;
              o[2] = (__bf16)v.z; o[3] = (__bf16)v.w; }
  }
}

// ---------------------------------------------------------------------------
// Weight pack: f32 row-major [Kdim x NcolSrc] -> bf16 B-fragment order,
// cols padded to NcolPad (multiple of 16) with zeros.
// ---------------------------------------------------------------------------
__global__ __launch_bounds__(256) void pack_kernel(const float* __restrict__ W,
                                                   __bf16* __restrict__ out,
                                                   int Kdim, int NcolPad, int NcolSrc,
                                                   int nMat) {
  size_t total = (size_t)nMat * NcolPad * Kdim;
  size_t t = (size_t)blockIdx.x * 256 + threadIdx.x;
  if (t >= total) return;
  int per = NcolPad * Kdim;
  int mi  = (int)(t / per);
  int r   = (int)(t % per);
  int e    = r & 15;
  int lane = (r >> 4) & 31;
  int rem  = r >> 9;
  int ksteps = Kdim >> 5;
  int s  = rem % ksteps;
  int ct = rem / ksteps;
  int col = ct * 16 + (lane & 15);
  int kk  = s * 32 + 8 * (lane >> 4) + (e & 7) + 16 * (e >> 3);
  float v = (col < NcolSrc)
                ? W[(size_t)mi * Kdim * NcolSrc + (size_t)kk * NcolSrc + col]
                : 0.f;
  out[t] = (__bf16)v;
}

// ---------------------------------------------------------------------------
// Stage 1: prediction heads.  One wave per (16-row tile, mode k).
// reg[n,k,:] = (relu(gn(relu(gn(x@W1))@W2) + x)) @ Wo + bo + ctr
// ---------------------------------------------------------------------------
__global__ __launch_bounds__(32) void stage1_kernel(
    const float* __restrict__ actors, const float* __restrict__ ctrs,
    const __bf16* __restrict__ w1p, const __bf16* __restrict__ w2p,
    const __bf16* __restrict__ wop,
    const float* __restrict__ g1, const float* __restrict__ b1,
    const float* __restrict__ g2, const float* __restrict__ b2,
    const float* __restrict__ bo, float* __restrict__ regOut) {
  __shared__ float  actF[16 * DM];
  __shared__ __bf16 actB[16 * DM];
  __shared__ float  H[16 * DM];
  __shared__ __bf16 Abuf[16 * DM];
  __shared__ float  cf[16][2];

  const int lane = threadIdx.x;
  const int n0   = blockIdx.x * 16;
  const int k    = blockIdx.y;

  load_act_tile<true, true>(actors, n0, lane, actF, actB);
  { int r = lane >> 1, xy = lane & 1;
    cf[r][xy] = ctrs[(size_t)(n0 + r) * 2 + xy]; }
  __syncthreads();

  const v8f z = {0.f, 0.f, 0.f, 0.f, 0.f, 0.f, 0.f, 0.f};

  // GEMM1: x @ W1[k]
  const v16bf* W1 = (const v16bf*)(w1p + (size_t)k * 16384);
#pragma unroll 1
  for (int c = 0; c < 8; ++c)
    store_tile(H, DM, c, lane, gemm_col<4>(actB, DM, W1, c, lane, z));
  __syncthreads();
  group_norm16<false, true, true, false>(H, g1 + k * DM, b1 + k * DM, nullptr,
                                         Abuf, nullptr, lane);
  __syncthreads();

  // GEMM2: h @ W2[k]; gn; +x; relu
  const v16bf* W2 = (const v16bf*)(w2p + (size_t)k * 16384);
#pragma unroll 1
  for (int c = 0; c < 8; ++c)
    store_tile(H, DM, c, lane, gemm_col<4>(Abuf, DM, W2, c, lane, z));
  __syncthreads();
  group_norm16<true, true, true, false>(H, g2 + k * DM, b2 + k * DM, actF,
                                        Abuf, nullptr, lane);
  __syncthreads();

  // GEMM3: h @ Wo[k]  (60 cols padded to 64) -> reg (+bo +ctr)
  const v16bf* Wo = (const v16bf*)(wop + (size_t)k * 8192);
#pragma unroll 1
  for (int c = 0; c < 4; ++c) {
    v8f acc = gemm_col<4>(Abuf, DM, Wo, c, lane, z);
    int o = c * 16 + (lane & 15);
    if (o < PO) {
      float bov = bo[k * PO + o];
      int r0 = 8 * (lane >> 4);
#pragma unroll
      for (int i = 0; i < 8; ++i) {
        int n = n0 + r0 + i;
        regOut[((size_t)n * KM + k) * PO + o] = acc[i] + bov + cf[r0 + i][o & 1];
      }
    }
  }
}

// ---------------------------------------------------------------------------
// Stage 2: destination attention + classification scores.
// ---------------------------------------------------------------------------
__global__ __launch_bounds__(32) void stage2_kernel(
    const float* __restrict__ actors, const float* __restrict__ ctrs,
    const float* __restrict__ regIn,
    const float* __restrict__ dW0, const float* __restrict__ db0,
    const __bf16* __restrict__ dW1p, const float* __restrict__ dg1,
    const float* __restrict__ db1,
    const __bf16* __restrict__ agtp, const float* __restrict__ ag,
    const float* __restrict__ ab,
    const __bf16* __restrict__ cw1p, const float* __restrict__ cg1,
    const float* __restrict__ cb1,
    const __bf16* __restrict__ cw2p, const float* __restrict__ cg2,
    const float* __restrict__ cb2,
    const float* __restrict__ cWo, const float* __restrict__ cbo,
    float* __restrict__ featsOut, float* __restrict__ clsOut) {
  __shared__ __bf16 actB[16 * DM];
  __shared__ __bf16 bufX[16 * DM];
  __shared__ __bf16 bufY[16 * DM];
  __shared__ float  H[16 * DM];
  __shared__ float  featF[16 * DM];
  __shared__ float  distS[16][2];

  const int lane = threadIdx.x;
  const int n0   = blockIdx.x * 16;
  const int k    = blockIdx.y;

  load_act_tile<false, true>(actors, n0, lane, nullptr, actB);
  { int r = lane >> 1, xy = lane & 1;
    size_t n = (size_t)(n0 + r);
    distS[r][xy] = ctrs[n * 2 + xy] - regIn[(n * KM + k) * PO + 58 + xy]; }
  __syncthreads();

  // dh0 = relu(dist @ dist_W0 + b0): K=2, do on VALU.
#pragma unroll 1
  for (int i = 0; i < 64; ++i) {
    int t = lane + 32 * i;
    int r = t >> 7, c = t & 127;
    float y = distS[r][0] * dW0[c] + distS[r][1] * dW0[DM + c] + db0[c];
    bufX[r * DM + c] = (__bf16)fmaxf(y, 0.f);
  }
  __syncthreads();

  const v8f z = {0.f, 0.f, 0.f, 0.f, 0.f, 0.f, 0.f, 0.f};

  // dh = relu(gn(dh0 @ dist_W1))
  const v16bf* DW1 = (const v16bf*)dW1p;
#pragma unroll 1
  for (int c = 0; c < 8; ++c)
    store_tile(H, DM, c, lane, gemm_col<4>(bufX, DM, DW1, c, lane, z));
  __syncthreads();
  group_norm16<false, true, true, false>(H, dg1, db1, nullptr, bufY, nullptr, lane);
  __syncthreads();

  // feats = relu(gn([dh | actors] @ agt_W)), K = 256 (8 k-steps).
  const v16bf* AG = (const v16bf*)agtp;
#pragma unroll 1
  for (int c = 0; c < 8; ++c) {
    v8f acc = z;
#pragma unroll
    for (int s = 0; s < 4; ++s)
      acc = wmma_bf16(load_a_frag(bufY, DM, lane, s), AG[(c * 8 + s) * 32 + lane], acc);
#pragma unroll
    for (int s = 0; s < 4; ++s)
      acc = wmma_bf16(load_a_frag(actB, DM, lane, s), AG[(c * 8 + 4 + s) * 32 + lane], acc);
    store_tile(H, DM, c, lane, acc);
  }
  __syncthreads();
  group_norm16<false, true, true, true>(H, ag, ab, nullptr, bufX, featF, lane);
  __syncthreads();

  // feats -> global (row n*K + k).
#pragma unroll 1
  for (int i = 0; i < 16; ++i) {
    int t = lane + 32 * i;
    int r = t >> 5, c = (t & 31) * 4;
    float4 v = *(const float4*)(featF + r * DM + c);
    *(float4*)(featsOut + ((size_t)(n0 + r) * KM + k) * DM + c) = v;
  }

  // cls LinearRes: relu(gn(f@W1)); gn(.@W2) + f; relu
  const v16bf* CW1 = (const v16bf*)cw1p;
#pragma unroll 1
  for (int c = 0; c < 8; ++c)
    store_tile(H, DM, c, lane, gemm_col<4>(bufX, DM, CW1, c, lane, z));
  __syncthreads();
  group_norm16<false, true, true, false>(H, cg1, cb1, nullptr, bufY, nullptr, lane);
  __syncthreads();

  const v16bf* CW2 = (const v16bf*)cw2p;
#pragma unroll 1
  for (int c = 0; c < 8; ++c)
    store_tile(H, DM, c, lane, gemm_col<4>(bufY, DM, CW2, c, lane, z));
  __syncthreads();
  group_norm16<true, true, false, true>(H, cg2, cb2, featF, nullptr, featF, lane);
  __syncthreads();

  // cls score = c . cls_Wo + cbo  (128-dot per row)
  const int c0 = lane * 4;
  float w4[4];
#pragma unroll
  for (int j = 0; j < 4; ++j) w4[j] = cWo[c0 + j];
  float cb = cbo[0];
#pragma unroll 1
  for (int r = 0; r < 16; ++r) {
    float4 xv = *(const float4*)(featF + r * DM + c0);
    float s = xv.x * w4[0] + xv.y * w4[1] + xv.z * w4[2] + xv.w * w4[3];
#pragma unroll
    for (int m = 16; m >= 1; m >>= 1) s += __shfl_xor(s, m, 32);
    if (lane == 0) clsOut[(size_t)(n0 + r) * KM + k] = s + cb;
  }
}

// ---------------------------------------------------------------------------
// Stage 3: softmax over K, stable descending argsort, in-place reg permute.
// One wave per row (8 waves / 256-thread block).
// ---------------------------------------------------------------------------
__global__ __launch_bounds__(256) void stage3_kernel(const float* __restrict__ clsIn,
                                                     float* __restrict__ clsSorted,
                                                     float* __restrict__ reg) {
  const int lane = threadIdx.x & 31;
  const int n    = blockIdx.x * 8 + (threadIdx.x >> 5);

  float p[KM];
  float mx = -1e30f;
#pragma unroll
  for (int j = 0; j < KM; ++j) { p[j] = clsIn[(size_t)n * KM + j]; mx = fmaxf(mx, p[j]); }
  float sum = 0.f;
#pragma unroll
  for (int j = 0; j < KM; ++j) { p[j] = __expf(p[j] - mx); sum += p[j]; }
  float rs = 1.f / sum;
#pragma unroll
  for (int j = 0; j < KM; ++j) p[j] *= rs;

  int order[KM];
  bool used[KM];
#pragma unroll
  for (int j = 0; j < KM; ++j) used[j] = false;
#pragma unroll
  for (int pos = 0; pos < KM; ++pos) {
    int best = 0; float bv = -1.f;
#pragma unroll
    for (int j = 0; j < KM; ++j)
      if (!used[j] && p[j] > bv) { bv = p[j]; best = j; }
    used[best] = true; order[pos] = best;
  }
  if (lane < KM) clsSorted[(size_t)n * KM + lane] = p[order[lane]];

  float* base = reg + (size_t)n * (KM * PO);
  float tmp[12];
  int cnt = 0;
  for (int f = lane; f < KM * PO; f += 32) {
    int j = f / PO, o = f - j * PO;
    tmp[cnt++] = base[order[j] * PO + o];
  }
  // all gathers must land before any lane overwrites the row
  asm volatile("s_wait_loadcnt 0x0" ::: "memory");
  cnt = 0;
  for (int f = lane; f < KM * PO; f += 32) base[f] = tmp[cnt++];
}

// ---------------------------------------------------------------------------
// Host-side launch
// ---------------------------------------------------------------------------
extern "C" void kernel_launch(void* const* d_in, const int* in_sizes, int n_in,
                              void* d_out, int out_size, void* d_ws, size_t ws_size,
                              hipStream_t stream) {
  (void)in_sizes; (void)n_in; (void)out_size; (void)ws_size;
  const float* actors  = (const float*)d_in[0];
  const float* ctrs    = (const float*)d_in[1];
  const float* pred_W1 = (const float*)d_in[2];
  const float* pred_g1 = (const float*)d_in[3];
  const float* pred_b1 = (const float*)d_in[4];
  const float* pred_W2 = (const float*)d_in[5];
  const float* pred_g2 = (const float*)d_in[6];
  const float* pred_b2 = (const float*)d_in[7];
  const float* pred_Wo = (const float*)d_in[8];
  const float* pred_bo = (const float*)d_in[9];
  const float* dist_W0 = (const float*)d_in[10];
  const float* dist_b0 = (const float*)d_in[11];
  const float* dist_W1 = (const float*)d_in[12];
  const float* dist_g1 = (const float*)d_in[13];
  const float* dist_b1 = (const float*)d_in[14];
  const float* agt_W   = (const float*)d_in[15];
  const float* agt_g   = (const float*)d_in[16];
  const float* agt_b   = (const float*)d_in[17];
  const float* cls_W1  = (const float*)d_in[18];
  const float* cls_g1  = (const float*)d_in[19];
  const float* cls_b1  = (const float*)d_in[20];
  const float* cls_W2  = (const float*)d_in[21];
  const float* cls_g2  = (const float*)d_in[22];
  const float* cls_b2  = (const float*)d_in[23];
  const float* cls_Wo  = (const float*)d_in[24];
  const float* cls_bo  = (const float*)d_in[25];

  const int N = 100000;
  // workspace layout (bf16 packed weights, then f32 cls scores)
  __bf16* w1p  = (__bf16*)d_ws;              // 6*128*128
  __bf16* w2p  = w1p  + 6 * 16384;
  __bf16* wop  = w2p  + 6 * 16384;           // 6*128*64 (padded)
  __bf16* dw1p = wop  + 6 * 8192;
  __bf16* agtp = dw1p + 16384;               // 256*128
  __bf16* cw1p = agtp + 32768;
  __bf16* cw2p = cw1p + 16384;
  float*  clsS = (float*)(cw2p + 16384);     // N*K scores

  float* out      = (float*)d_out;
  float* outCls   = out;                              // N*K
  float* outReg   = out + (size_t)N * KM;             // N*K*PO
  float* outFeats = outReg + (size_t)N * KM * PO;     // N*K*DM

  auto pack = [&](const float* W, __bf16* o, int Kd, int Np, int Ns, int nM) {
    size_t total = (size_t)nM * Np * Kd;
    pack_kernel<<<dim3((unsigned)((total + 255) / 256)), 256, 0, stream>>>(
        W, o, Kd, Np, Ns, nM);
  };
  pack(pred_W1, w1p, 128, 128, 128, 6);
  pack(pred_W2, w2p, 128, 128, 128, 6);
  pack(pred_Wo, wop, 128, 64, 60, 6);
  pack(dist_W1, dw1p, 128, 128, 128, 1);
  pack(agt_W,   agtp, 256, 128, 128, 1);
  pack(cls_W1,  cw1p, 128, 128, 128, 1);
  pack(cls_W2,  cw2p, 128, 128, 128, 1);

  dim3 grid(N / 16, KM);
  stage1_kernel<<<grid, 32, 0, stream>>>(actors, ctrs, w1p, w2p, wop,
                                         pred_g1, pred_b1, pred_g2, pred_b2,
                                         pred_bo, outReg);
  stage2_kernel<<<grid, 32, 0, stream>>>(actors, ctrs, outReg,
                                         dist_W0, dist_b0,
                                         dw1p, dist_g1, dist_b1,
                                         agtp, agt_g, agt_b,
                                         cw1p, cls_g1, cls_b1,
                                         cw2p, cls_g2, cls_b2,
                                         cls_Wo, cls_bo,
                                         outFeats, clsS);
  stage3_kernel<<<N / 8, 256, 0, stream>>>(clsS, outCls, outReg);
}
